// TimeMixer_51728586113084
// MI455X (gfx1250) — matrix operand order repaired and checked by reading, not compiled, not run on hardware
//
#include <hip/hip_runtime.h>
#include <hip/hip_bf16.h>

typedef __attribute__((ext_vector_type(16))) __bf16 v16bf;
typedef __attribute__((ext_vector_type(8)))  float  v8f;

#define B_   4
#define T_   1024
#define C_   1024
#define H_   16
#define BT_  4096
#define EPS_ 0.00064f

#define USE_ASYNC 1

// LDS byte offset of a __shared__ object (addrspacecast to LDS, then ptrtoint)
__device__ __forceinline__ unsigned lds_off_of(const void* p) {
    return (unsigned)(unsigned long long)(__attribute__((address_space(3))) const char*)p;
}

// 16B async global->LDS copy, tracked by ASYNCcnt
__device__ __forceinline__ void async_copy16(unsigned ldsoff, const float* gaddr) {
#if USE_ASYNC
    asm volatile("global_load_async_to_lds_b128 %0, %1, off"
                 :: "v"(ldsoff), "v"(gaddr) : "memory");
#endif
}

__device__ __forceinline__ void wait_asynccnt0() {
#if USE_ASYNC
    asm volatile("s_wait_asynccnt 0" ::: "memory");
#endif
}

// ---------------------------------------------------------------------------
// Kernel 1: token shift + maa LoRA expansion -> xm[4] = {xrg, xwa, xk, xv}
// ---------------------------------------------------------------------------
__global__ __launch_bounds__(256)
void token_shift_kernel(const float* __restrict__ x,
                        const float* __restrict__ time_maa_x,
                        const float* __restrict__ time_maa,   // [4][C]
                        const float* __restrict__ maa_w1,     // [128][C]
                        const float* __restrict__ maa_w2,     // [4][32][C]
                        float* __restrict__ xm)               // 4 buffers, stride BT_*C_
{
    __shared__ float xs[C_], xxs[C_], mixs[C_], part[256], lo[128];
    const int tid = threadIdx.x;
    const int bt  = blockIdx.x;
    const int t   = bt & (T_ - 1);
    const size_t row = (size_t)bt * C_;

#pragma unroll
    for (int i = 0; i < 4; ++i) {
        int c = tid * 4 + i;
        float xv = x[row + c];
        float xp = (t == 0) ? 0.0f : x[row - C_ + c];
        float dx = xp - xv;
        xs[c] = xv; xxs[c] = dx;
        mixs[c] = xv + dx * time_maa_x[c];
    }
    __syncthreads();

    { // lo = tanh(mix @ maa_w1.T): 128 outputs, 2 threads per output
        int d = tid >> 1, h = tid & 1;
        const float* wrow = maa_w1 + (size_t)d * C_ + h * 512;
        const float* mrow = mixs + h * 512;
        float s = 0.f;
#pragma unroll 8
        for (int c = 0; c < 512; ++c) s += mrow[c] * wrow[c];
        part[tid] = s;
    }
    __syncthreads();
    if (tid < 128) lo[tid] = tanhf(part[2 * tid] + part[2 * tid + 1]);
    __syncthreads();

#pragma unroll
    for (int i = 0; i < 4; ++i) {
        int c = tid * 4 + i;
        float xv = xs[c], dx = xxs[c];
#pragma unroll
        for (int g = 0; g < 4; ++g) {
            float acc = time_maa[g * C_ + c];
#pragma unroll
            for (int j = 0; j < 32; ++j)
                acc += lo[g * 32 + j] * maa_w2[(size_t)(g * 32 + j) * C_ + c];
            xm[(size_t)g * ((size_t)BT_ * C_) + row + c] = xv + dx * acc;
        }
    }
}

// ---------------------------------------------------------------------------
// WMMA GEMM, templated: Cmat[M,N] = act(A[M,K]) @ B
//   TRANSW=1 : B(k,n) = W[n*K+k]  (C = A @ W^T)   -> async LDS staging
//   TRANSW=0 : B(k,n) = W[k*N+n]                  -> sync staging
//   TANHA=1  : tanh applied to A during staging (forces sync A staging)
// Block tile 128 x TNP, 8 waves, wave tile 32 x (TNP/2), K-step 32.
// fp32 tiles in LDS, converted to bf16 during fragment gather.
// ---------------------------------------------------------------------------
#define TM 128
#define TK 32
#define LDP 36   // padded row stride (floats); 144B = 9*16B keeps 16B chunks aligned

template<int TNP, int TRANSW, int TANHA>
__global__ __launch_bounds__(256)
void wmma_gemm_kernel(const float* __restrict__ A, const float* __restrict__ W,
                      float* __restrict__ Cmat, int M, int N, int K)
{
    constexpr int JN = TNP / 32;            // N subtiles per wave
    __shared__ __align__(16) float As[TM][LDP];
    __shared__ __align__(16) float Bs[TNP][LDP];

    const int tid  = threadIdx.x;
    const int wave = tid >> 5;
    const int lane = tid & 31;
    const int wm   = wave & 3;              // M sub-block: wm*32
    const int wn   = wave >> 2;             // N sub-block: wn*(TNP/2)
    const int m0   = blockIdx.y * TM;
    const int n0   = blockIdx.x * TNP;

    const int frag  = lane & 15;
    const int khalf = lane >> 4;

    const unsigned asBase = lds_off_of(&As[0][0]);
    const unsigned bsBase = lds_off_of(&Bs[0][0]);

    v8f acc[2][JN];
#pragma unroll
    for (int i = 0; i < 2; ++i)
#pragma unroll
        for (int j = 0; j < JN; ++j)
#pragma unroll
            for (int e = 0; e < 8; ++e) acc[i][j][e] = 0.0f;

    for (int k0 = 0; k0 < K; k0 += TK) {
        if constexpr (USE_ASYNC && TRANSW == 1 && TANHA == 0) {
            // A tile: 128x32 f32 = 1024 16B chunks
#pragma unroll
            for (int it = 0; it < 4; ++it) {
                int idx = tid + it * 256;
                int r = idx >> 3, c4 = idx & 7;
                async_copy16(asBase + (unsigned)(r * LDP + c4 * 4) * 4u,
                             A + (size_t)(m0 + r) * K + (k0 + c4 * 4));
            }
            // B tile (row-major N x K): TNP*8 chunks
#pragma unroll
            for (int it = 0; it < TNP / 32; ++it) {
                int idx = tid + it * 256;
                int n = idx >> 3, c4 = idx & 7;
                async_copy16(bsBase + (unsigned)(n * LDP + c4 * 4) * 4u,
                             W + (size_t)(n0 + n) * K + (k0 + c4 * 4));
            }
            wait_asynccnt0();
        } else {
            // synchronous staging (tanh / non-transposed W)
#pragma unroll
            for (int it = 0; it < 16; ++it) {
                int idx = tid + it * 256;
                int r = idx >> 5, c = idx & 31;
                float va = A[(size_t)(m0 + r) * K + (k0 + c)];
                if (TANHA) va = tanhf(va);
                As[r][c] = va;
            }
#pragma unroll
            for (int it = 0; it < TNP / 8; ++it) {
                int idx = tid + it * 256;
                int n = idx >> 5, c = idx & 31;
                float vb = TRANSW ? W[(size_t)(n0 + n) * K + (k0 + c)]
                                  : W[(size_t)(k0 + c) * N + (n0 + n)];
                Bs[n][c] = vb;
            }
            if (k0 + TK < K)
                __builtin_prefetch(A + (size_t)(m0 + (tid >> 1)) * K + (k0 + TK), 0, 1);
        }
        __syncthreads();

        // fragment gather (ISA 7.12.2 layouts), f32 -> bf16
        v16bf af[2], bfr[JN];
#pragma unroll
        for (int i = 0; i < 2; ++i) {
            int mr = wm * 32 + i * 16 + frag;
#pragma unroll
            for (int e = 0; e < 8; ++e) af[i][e]     = (__bf16)As[mr][khalf * 8 + e];
#pragma unroll
            for (int e = 0; e < 8; ++e) af[i][8 + e] = (__bf16)As[mr][16 + khalf * 8 + e];
        }
#pragma unroll
        for (int j = 0; j < JN; ++j) {
            int nc = wn * (TNP / 2) + j * 16 + frag;
#pragma unroll
            for (int e = 0; e < 8; ++e) bfr[j][e]     = (__bf16)Bs[nc][khalf * 8 + e];
#pragma unroll
            for (int e = 0; e < 8; ++e) bfr[j][8 + e] = (__bf16)Bs[nc][16 + khalf * 8 + e];
        }
#pragma unroll
        for (int i = 0; i < 2; ++i)
#pragma unroll
            for (int j = 0; j < JN; ++j)
                acc[i][j] = __builtin_amdgcn_wmma_f32_16x16x32_bf16(
                    false, af[i], false, bfr[j], (short)0, acc[i][j], false, false);
        __syncthreads();
    }

    // store: VGPR r -> M = mbase + r; N = lane%16
    const int nst   = lane & 15;
    const int mbase = (lane >> 4) * 8;
#pragma unroll
    for (int i = 0; i < 2; ++i)
#pragma unroll
        for (int j = 0; j < JN; ++j) {
            int gm = m0 + wm * 32 + i * 16 + mbase;
            int gn = n0 + wn * (TNP / 2) + j * 16 + nst;
#pragma unroll
            for (int r = 0; r < 8; ++r)
                Cmat[(size_t)(gm + r) * N + gn] = acc[i][j][r];
        }
}

// ---------------------------------------------------------------------------
// Kernel 3: fused rank-16 LoRAs + decay/k/kk finalize. One block per token.
// kkbuf may alias xwa, abuf may alias xk (row read to LDS before writes).
// ---------------------------------------------------------------------------
__global__ __launch_bounds__(256)
void lora_small_kernel(const float* xwa, const float* xk,
                       const float* __restrict__ aaa_w1, const float* __restrict__ aaa_w2,
                       const float* __restrict__ kkk_w1, const float* __restrict__ kkk_w2,
                       const float* __restrict__ ma_w1,  const float* __restrict__ ma_w2,
                       const float* __restrict__ mk_w1,  const float* __restrict__ mk_w2,
                       const float* __restrict__ time_decay,
                       const float* __restrict__ time_aaaaa,
                       const float* __restrict__ time_misc_a,
                       const float* __restrict__ time_misc_k,
                       float* kbuf,   // in: k0 ; out: final k
                       float* wbuf,   // in: w2 lora ; out: wt = exp(-exp(w))
                       float* kkbuf,  // out: normalized kk
                       float* abuf)   // out: a
{
    __shared__ float xwas[C_], xks[C_], part[256], h[64], kks[C_], red[256], hn[H_];
    const int tid = threadIdx.x;
    const size_t row = (size_t)blockIdx.x * C_;

#pragma unroll
    for (int i = 0; i < 4; ++i) {
        int c = tid * 4 + i;
        xwas[c] = xwa[row + c];
        xks[c]  = xk[row + c];
    }
    __syncthreads();

    { // 64 rank-16 projections of length 1024; 4 threads per output
        int d = tid >> 2, q = tid & 3;
        const float *src, *wrow;
        if (d < 16)      { src = xwas; wrow = aaa_w1 + (size_t)d * C_; }
        else if (d < 32) { src = xwas; wrow = ma_w1  + (size_t)(d - 16) * C_; }
        else if (d < 48) { src = xks;  wrow = kkk_w1 + (size_t)(d - 32) * C_; }
        else             { src = xks;  wrow = mk_w1  + (size_t)(d - 48) * C_; }
        float s = 0.f;
        for (int c = q * 256; c < q * 256 + 256; ++c) s += src[c] * wrow[c];
        part[tid] = s;
    }
    __syncthreads();
    if (tid < 64) {
        float s = part[4 * tid] + part[4 * tid + 1] + part[4 * tid + 2] + part[4 * tid + 3];
        h[tid] = (tid >= 32 && tid < 48) ? tanhf(s) : s;  // tanh only for kkk branch
    }
    __syncthreads();

    float localss = 0.f;
#pragma unroll
    for (int i = 0; i < 4; ++i) {
        int c = tid * 4 + i;
        float acc_a = time_aaaaa[c], acc_ma = time_misc_a[c];
        float acc_kk = 0.f,          acc_mk = time_misc_k[c];
#pragma unroll
        for (int j = 0; j < 16; ++j) {
            acc_a  += h[j]      * aaa_w2[(size_t)j * C_ + c];
            acc_ma += h[16 + j] * ma_w2 [(size_t)j * C_ + c];
            acc_kk += h[32 + j] * kkk_w2[(size_t)j * C_ + c];
            acc_mk += h[48 + j] * mk_w2 [(size_t)j * C_ + c];
        }
        float a_c  = 1.f / (1.f + expf(-acc_a));
        float ma_c = 1.f / (1.f + expf(-acc_ma));
        float mk_c = 1.f / (1.f + expf(-acc_mk));
        float k0   = kbuf[row + c];
        float kkv  = k0 + acc_kk;
        kks[c] = kkv;
        localss += kkv * kkv;
        float u  = time_decay[c] + wbuf[row + c];
        float wv = -log1pf(expf(-u)) - 0.5f;          // -softplus(-u) - 0.5
        float kn = k0 * ma_c + k0 * a_c * (1.f - ma_c);
        kn *= expf(wv * mk_c);
        kbuf[row + c] = kn;
        wbuf[row + c] = expf(-expf(wv));              // wt for recurrence
        abuf[row + c] = a_c;
    }
    red[tid] = localss;
    __syncthreads();
    if (tid < H_) {
        float s = 0.f;
        for (int q = 0; q < 16; ++q) s += red[tid * 16 + q];
        hn[tid] = fmaxf(sqrtf(s), 1e-12f);
    }
    __syncthreads();
#pragma unroll
    for (int i = 0; i < 4; ++i) {
        int c = tid * 4 + i;
        kkbuf[row + c] = kks[c] / hn[c >> 6];
    }
}

// ---------------------------------------------------------------------------
// Kernel 4: delta-rule recurrence. One block (64 thr) per (b,h); thread i owns
// state row i (64 f32 in VGPRs), per-step vectors broadcast through LDS.
// ---------------------------------------------------------------------------
__global__ __launch_bounds__(64)
void recurrence_kernel(const float* __restrict__ rbuf, const float* __restrict__ wbuf,
                       const float* __restrict__ kbuf, const float* __restrict__ vbuf,
                       const float* __restrict__ kkbuf, const float* __restrict__ abuf,
                       float* __restrict__ ybuf)
{
    __shared__ float qv[64], wv[64], kv[64], vv[64], av[64], bv[64];
    const int i  = threadIdx.x;
    const int b  = blockIdx.x >> 4;
    const int hh = blockIdx.x & (H_ - 1);

    float s[64];
#pragma unroll
    for (int j = 0; j < 64; ++j) s[j] = 0.f;

    for (int t = 0; t < T_; ++t) {
        size_t base = ((size_t)(b * T_ + t)) * C_ + hh * 64;
        qv[i] = rbuf[base + i];
        wv[i] = wbuf[base + i];
        kv[i] = kbuf[base + i];
        vv[i] = vbuf[base + i];
        float kkv = kkbuf[base + i];
        float a_  = abuf[base + i];
        av[i] = kkv;
        bv[i] = -kkv * a_;
        __syncthreads();

        float sa = 0.f;
#pragma unroll
        for (int j = 0; j < 64; ++j) sa += s[j] * av[j];
        float vi = vv[i], y = 0.f;
#pragma unroll
        for (int j = 0; j < 64; ++j) {
            s[j] = s[j] * wv[j] + sa * bv[j] + vi * kv[j];
            y += s[j] * qv[j];
        }
        ybuf[base + i] = y;
        __syncthreads();
    }
}

// ---------------------------------------------------------------------------
// Kernel 5: GroupNorm (H groups) + per-head bonus + gate. One block per token.
// ---------------------------------------------------------------------------
__global__ __launch_bounds__(256)
void post_kernel(const float* __restrict__ ybuf, const float* __restrict__ rbuf,
                 const float* __restrict__ kbuf, const float* __restrict__ vbuf,
                 const float* __restrict__ gbuf, const float* __restrict__ ln_w,
                 const float* __restrict__ ln_b, const float* __restrict__ faaaa,
                 float* ygbuf)
{
    __shared__ float red0[256], red1[256], red2[256], mu[H_], rs[H_], rk[H_];
    const int tid = threadIdx.x;
    const size_t row = (size_t)blockIdx.x * C_;

    float yv[4], s0 = 0.f, s1 = 0.f, s2 = 0.f;
#pragma unroll
    for (int i = 0; i < 4; ++i) {
        int c = tid * 4 + i;
        float y = ybuf[row + c];
        yv[i] = y;
        s0 += y; s1 += y * y;
        s2 += rbuf[row + c] * kbuf[row + c] * faaaa[c];
    }
    red0[tid] = s0; red1[tid] = s1; red2[tid] = s2;
    __syncthreads();
    if (tid < H_) {
        float a = 0.f, bq = 0.f, cq = 0.f;
        for (int q = 0; q < 16; ++q) {
            a += red0[tid * 16 + q]; bq += red1[tid * 16 + q]; cq += red2[tid * 16 + q];
        }
        float m = a * (1.f / 64.f);
        mu[tid] = m;
        rs[tid] = rsqrtf(bq * (1.f / 64.f) - m * m + EPS_);
        rk[tid] = cq;
    }
    __syncthreads();
#pragma unroll
    for (int i = 0; i < 4; ++i) {
        int c = tid * 4 + i, hd = c >> 6;
        float gn   = (yv[i] - mu[hd]) * rs[hd] * ln_w[c] + ln_b[c];
        float yfin = gn + rk[hd] * vbuf[row + c];
        ygbuf[row + c] = yfin * gbuf[row + c];
    }
}

// ---------------------------------------------------------------------------
extern "C" void kernel_launch(void* const* d_in, const int* in_sizes, int n_in,
                              void* d_out, int out_size, void* d_ws, size_t ws_size,
                              hipStream_t stream) {
    const float* x           = (const float*)d_in[0];
    const float* time_maa_x  = (const float*)d_in[1];
    const float* time_maa    = (const float*)d_in[2];
    const float* maa_w1      = (const float*)d_in[3];
    const float* maa_w2      = (const float*)d_in[4];
    const float* decay_w1    = (const float*)d_in[5];
    const float* decay_w2    = (const float*)d_in[6];
    const float* aaa_w1      = (const float*)d_in[7];
    const float* aaa_w2      = (const float*)d_in[8];
    const float* kkk_w1      = (const float*)d_in[9];
    const float* kkk_w2      = (const float*)d_in[10];
    const float* gate_w1     = (const float*)d_in[11];
    const float* gate_w2     = (const float*)d_in[12];
    const float* ma_w1       = (const float*)d_in[13];
    const float* ma_w2       = (const float*)d_in[14];
    const float* mk_w1       = (const float*)d_in[15];
    const float* mk_w2       = (const float*)d_in[16];
    const float* time_decay  = (const float*)d_in[17];
    const float* time_faaaa  = (const float*)d_in[18];
    const float* time_aaaaa  = (const float*)d_in[19];
    const float* time_misc_a = (const float*)d_in[20];
    const float* time_misc_k = (const float*)d_in[21];
    const float* Wr          = (const float*)d_in[22];
    const float* Wk          = (const float*)d_in[23];
    const float* Wv          = (const float*)d_in[24];
    const float* Wo          = (const float*)d_in[25];
    const float* ln_w        = (const float*)d_in[26];
    const float* ln_b        = (const float*)d_in[27];
    float* out = (float*)d_out;

    float* ws = (float*)d_ws;
    const size_t S = (size_t)BT_ * C_;
    float* xrg = ws + 0 * S;           // later reused as y*g
    float* xwa = ws + 1 * S;           // later reused as kk
    float* xk  = ws + 2 * S;           // later reused as a
    float* xv  = ws + 3 * S;           // later reused as y
    float* rb  = ws + 4 * S;
    float* kb  = ws + 5 * S;
    float* vb  = ws + 6 * S;
    float* gb  = ws + 7 * S;
    float* wb  = ws + 8 * S;
    float* hgate  = ws + 9 * S;        // [BT,128]
    float* hdecay = hgate + (size_t)BT_ * 128;  // [BT,64]
    float* kkb = xwa;
    float* ab  = xk;
    float* yb  = xv;
    float* yg  = xrg;

    dim3 blk(256);
    dim3 gBig(C_ / 128, BT_ / TM);     // (8, 32)

    // Stage A: token shift + maa LoRA -> xrg,xwa,xk,xv
    token_shift_kernel<<<BT_, blk, 0, stream>>>(x, time_maa_x, time_maa, maa_w1, maa_w2, xrg);

    // Stage B: big WMMA GEMMs (async-staged, C = A @ W^T)
    wmma_gemm_kernel<128, 1, 0><<<gBig, blk, 0, stream>>>(xrg, Wr, rb, BT_, C_, C_);
    wmma_gemm_kernel<128, 1, 0><<<gBig, blk, 0, stream>>>(xk,  Wk, kb, BT_, C_, C_);
    wmma_gemm_kernel<128, 1, 0><<<gBig, blk, 0, stream>>>(xv,  Wv, vb, BT_, C_, C_);
    wmma_gemm_kernel<128, 1, 0><<<dim3(1, BT_ / TM), blk, 0, stream>>>(
        xrg, gate_w1, hgate, BT_, 128, C_);
    wmma_gemm_kernel<128, 0, 1><<<gBig, blk, 0, stream>>>(hgate, gate_w2, gb, BT_, C_, 128);
    wmma_gemm_kernel<64, 1, 0><<<dim3(1, BT_ / TM), blk, 0, stream>>>(
        xwa, decay_w1, hdecay, BT_, 64, C_);
    wmma_gemm_kernel<128, 0, 1><<<gBig, blk, 0, stream>>>(hdecay, decay_w2, wb, BT_, C_, 64);

    // Stage C: fused small LoRAs + decay/k/kk finalize
    lora_small_kernel<<<BT_, blk, 0, stream>>>(
        xwa, xk, aaa_w1, aaa_w2, kkk_w1, kkk_w2, ma_w1, ma_w2, mk_w1, mk_w2,
        time_decay, time_aaaaa, time_misc_a, time_misc_k, kb, wb, kkb, ab);

    // Stage D: delta-rule recurrence
    recurrence_kernel<<<B_ * H_, dim3(64), 0, stream>>>(rb, wb, kb, vb, kkb, ab, yb);

    // Stage E: GroupNorm + bonus + gate
    post_kernel<<<BT_, blk, 0, stream>>>(yb, rb, kb, vb, gb, ln_w, ln_b, time_faaaa, yg);

    // Stage F: output projection -> d_out
    wmma_gemm_kernel<128, 1, 0><<<gBig, blk, 0, stream>>>(yg, Wo, out, BT_, C_, C_);
}
